// CrossTransformer_27522150432886
// MI455X (gfx1250) — matrix operand order repaired and checked by reading, not compiled, not run on hardware
//
#include <hip/hip_runtime.h>
#include <hip/hip_bf16.h>

// ---------------------------------------------------------------------------
// CrossTransformer block for MI455X (gfx1250, wave32, WMMA).
// GEMMs: v_wmma_f32_16x16x32_f16, double-buffered LDS tiles, async-to-LDS
// staging (ASYNCcnt). Attention: flash-style with WMMA QK/PV.
// ---------------------------------------------------------------------------

typedef __attribute__((ext_vector_type(16))) _Float16 v16h;
typedef __attribute__((ext_vector_type(8)))  _Float16 v8h;
typedef __attribute__((ext_vector_type(8)))  float    v8f;
typedef __attribute__((ext_vector_type(4)))  int      v4i;

#define DMODEL 1024
#define NROWS  4096   // B*T = 4*1024
#define NHEAD  16
#define HD     64
#define BATCH  4
#define SEQ    1024

#ifndef __has_builtin
#define __has_builtin(x) 0
#endif
#if __has_builtin(__builtin_amdgcn_global_load_async_to_lds_b128)
#define HAVE_ASYNC_LDS 1
#else
#define HAVE_ASYNC_LDS 0
#endif

// 16-byte global -> LDS copy; async (ASYNCcnt-tracked) when the toolchain
// exposes the gfx1250 builtin, otherwise synchronous load+ds_store.
__device__ __forceinline__ void copy16_g2l(const _Float16* g, _Float16* l) {
#if HAVE_ASYNC_LDS
  __builtin_amdgcn_global_load_async_to_lds_b128(
      (__attribute__((address_space(1))) v4i*)g,
      (__attribute__((address_space(3))) v4i*)l, 0, 0);
#else
  *(v8h*)l = *(const v8h*)g;
#endif
}

__device__ __forceinline__ void wait_async_lds() {
#if HAVE_ASYNC_LDS
  asm volatile("s_wait_asynccnt 0" ::: "memory");
#endif
}

// ---- WMMA fragment loaders (wave32, 16x16x32 f16) -------------------------
// A fragment (16xK=32): lanes 0-15 row m=lane hold K {0..7,16..23};
//                       lanes 16-31 row m=lane-16 hold K {8..15,24..31}.
__device__ __forceinline__ v16h load_fragA(const _Float16* p, int ld) {
  int lane = threadIdx.x & 31;
  int m    = lane & 15;
  int kb   = (lane >> 4) * 8;
  const _Float16* row = p + (size_t)m * ld + kb;
  v8h a0 = *(const v8h*)(row);
  v8h a1 = *(const v8h*)(row + 16);
  v16h r;
#pragma unroll
  for (int i = 0; i < 8; ++i) { r[i] = a0[i]; r[8 + i] = a1[i]; }
  return r;
}

// B fragment (K=32 x 16), stored K-contiguous per column:
// lanes 0-15 col n=lane hold K {0..7,16..23}; lanes 16-31 col n=lane-16 hold K {8..15,24..31}.
__device__ __forceinline__ v16h load_fragB(const _Float16* p, int ld) {
  int lane = threadIdx.x & 31;
  int n    = lane & 15;
  int kb   = (lane >> 4) * 8;
  const _Float16* col = p + (size_t)n * ld + kb;
  v8h b0 = *(const v8h*)(col);
  v8h b1 = *(const v8h*)(col + 16);
  v16h r;
#pragma unroll
  for (int i = 0; i < 8; ++i) { r[i] = b0[i]; r[8 + i] = b1[i]; }
  return r;
}

// ---- helpers ---------------------------------------------------------------
__global__ __launch_bounds__(256) void f32_to_f16_kernel(
    const float* __restrict__ src, _Float16* __restrict__ dst, int n) {
  for (int i = blockIdx.x * 256 + threadIdx.x; i < n; i += gridDim.x * 256)
    dst[i] = (_Float16)src[i];
}

// row-wise LayerNorm + ReLU -> f16  (one block per row, 256 threads)
__global__ __launch_bounds__(256) void ln_relu_f16_kernel(
    const float* __restrict__ x, const float* __restrict__ g,
    const float* __restrict__ b, _Float16* __restrict__ out, int Dd) {
  __shared__ float rs[256], rs2[256];
  int row = blockIdx.x;
  const float* xr = x + (size_t)row * Dd;
  float s = 0.f, s2 = 0.f;
  for (int i = threadIdx.x; i < Dd; i += 256) {
    float v = xr[i];
    s += v; s2 += v * v;
  }
  rs[threadIdx.x] = s; rs2[threadIdx.x] = s2;
  __syncthreads();
  for (int st = 128; st > 0; st >>= 1) {
    if (threadIdx.x < st) {
      rs[threadIdx.x]  += rs[threadIdx.x + st];
      rs2[threadIdx.x] += rs2[threadIdx.x + st];
    }
    __syncthreads();
  }
  float mean = rs[0] / (float)Dd;
  float var  = rs2[0] / (float)Dd - mean * mean;
  float inv  = rsqrtf(var + 1e-5f);
  _Float16* orow = out + (size_t)row * Dd;
  for (int i = threadIdx.x; i < Dd; i += 256) {
    float v = (xr[i] - mean) * inv * g[i] + b[i];
    orow[i] = (_Float16)fmaxf(v, 0.f);
  }
}

// ---- Tiled WMMA GEMM:  C[M,N] = A[M,K](f16) @ B[K,N](f16) + bias (+res) ----
// Block tile 128x128, 8 waves (4x2), each wave: 2x4 tiles of 16x16.
// Double-buffered LDS; A staged via async-to-LDS, B transposed via registers.
// EPI: 0 = store f16 (bias); 1 = store f32 (bias + f32 residual); 2 = f32 (bias)
template <int EPI>
__global__ __launch_bounds__(256) void gemm_wmma_kernel(
    const _Float16* __restrict__ A, const _Float16* __restrict__ Bw,
    const float* __restrict__ bias, const float* __restrict__ resid,
    float* __restrict__ outF32, _Float16* __restrict__ outF16,
    int M, int N, int K) {
  __shared__ _Float16 sA[2][128 * 32];  // [m][k], k contiguous
  __shared__ _Float16 sB[2][128 * 32];  // [n][k], k contiguous (transposed)
  int bm = blockIdx.x * 128, bn = blockIdx.y * 128;
  int tid = threadIdx.x, wave = tid >> 5, lane = tid & 31;
  int wm = (wave >> 1) * 32;   // wave row offset in block tile
  int wn = (wave & 1) * 64;    // wave col offset
  v8f acc[2][4] = {};
  v8h breg[2];

  auto stageA = [&](int k0, int buf) {
#pragma unroll
    for (int i = 0; i < 2; ++i) {
      int c = tid + i * 256;                 // 512 chunks of 8 halves
      int r = c >> 2, cc = (c & 3) * 8;
      copy16_g2l(A + (size_t)(bm + r) * K + k0 + cc, &sA[buf][r * 32 + cc]);
    }
  };
  auto loadB = [&](int k0) {
#pragma unroll
    for (int i = 0; i < 2; ++i) {
      int c = tid + i * 256;
      int bk = c >> 4, bn8 = (c & 15) * 8;
      breg[i] = *(const v8h*)(Bw + (size_t)(k0 + bk) * N + bn + bn8);
    }
  };
  auto storeB = [&](int buf) {
#pragma unroll
    for (int i = 0; i < 2; ++i) {
      int c = tid + i * 256;
      int bk = c >> 4, bn8 = (c & 15) * 8;
#pragma unroll
      for (int j = 0; j < 8; ++j) sB[buf][(bn8 + j) * 32 + bk] = breg[i][j];
    }
  };

  // prologue: fill buffer 0
  stageA(0, 0);
  loadB(0);
  storeB(0);
  wait_async_lds();
  __syncthreads();

  for (int k0 = 0; k0 < K; k0 += 32) {
    int cur = (k0 >> 5) & 1, nxt = cur ^ 1;
    bool more = (k0 + 32 < K);
    if (more) {      // overlap next tile's traffic with this tile's WMMAs
      stageA(k0 + 32, nxt);
      loadB(k0 + 32);
    }
    v16h af[2], bf[4];
    af[0] = load_fragA(&sA[cur][(wm + 0) * 32], 32);
    af[1] = load_fragA(&sA[cur][(wm + 16) * 32], 32);
#pragma unroll
    for (int j = 0; j < 4; ++j) bf[j] = load_fragB(&sB[cur][(wn + j * 16) * 32], 32);
#pragma unroll
    for (int i = 0; i < 2; ++i)
#pragma unroll
      for (int j = 0; j < 4; ++j)
        acc[i][j] = __builtin_amdgcn_wmma_f32_16x16x32_f16(
            false, af[i], false, bf[j], (short)0, acc[i][j], false, false);
    if (more) storeB(nxt);
    wait_async_lds();
    __syncthreads();
  }

  // epilogue: C/D layout -> lane l, vgpr r: (m = r + 8*(l>>4), n = l&15)
  int cn = lane & 15, mh = (lane >> 4) * 8;
#pragma unroll
  for (int i = 0; i < 2; ++i)
#pragma unroll
    for (int j = 0; j < 4; ++j) {
      int col = bn + wn + j * 16 + cn;
      float bvv = bias[col];
#pragma unroll
      for (int r = 0; r < 8; ++r) {
        int row = bm + wm + i * 16 + mh + r;
        float v = acc[i][j][r] + bvv;
        if (EPI == 1) v += resid[(size_t)row * N + col];
        if (EPI == 0) outF16[(size_t)row * N + col] = (_Float16)v;
        else          outF32[(size_t)row * N + col] = v;
      }
    }
}

// ---- Flash attention: block = (b, h, 128 q rows), 8 waves x 16 q rows -----
__global__ __launch_bounds__(256) void attn_kernel(
    const _Float16* __restrict__ Q, const _Float16* __restrict__ Kf,
    const _Float16* __restrict__ Vf, const unsigned char* __restrict__ mask,
    _Float16* __restrict__ O) {
  __shared__ _Float16 sK[32 * 64];       // [key][d]
  __shared__ _Float16 sVt[64 * 32];      // [dv][key]
  __shared__ _Float16 sP[8 * 16 * 32];   // per-wave P staging (16 q x 32 keys)
  __shared__ unsigned char sMask[SEQ];   // key mask for this batch

  int b  = blockIdx.y >> 4;
  int h  = blockIdx.y & 15;
  int q0 = blockIdx.x * 128;
  int tid = threadIdx.x, wave = tid >> 5, lane = tid & 31;
  int cn = lane & 15, hi = lane >> 4;

  // stage the whole mask row once (1 KB); first loop barrier publishes it
  *(unsigned int*)(sMask + tid * 4) =
      *(const unsigned int*)(mask + b * SEQ + tid * 4);

  // Q fragments for this wave's 16 query rows (d split 0..31 / 32..63)
  const _Float16* qbase =
      Q + ((size_t)(b * SEQ) + q0 + wave * 16) * DMODEL + h * HD;
  v16h aq[2];
  aq[0] = load_fragA(qbase, DMODEL);
  aq[1] = load_fragA(qbase + 32, DMODEL);

  v8f acc[4] = {};
  float mrun[8], lrun[8];
#pragma unroll
  for (int r = 0; r < 8; ++r) { mrun[r] = -3.0e38f; lrun[r] = 0.f; }

  for (int kc = 0; kc < SEQ; kc += 32) {
    __syncthreads();  // protect previous iteration's sK/sVt reads
    {  // stage K chunk [32 keys][64 d] (async) and V transposed [64 dv][32 keys]
      int key = tid >> 3, d8 = (tid & 7) * 8;
      const _Float16* kr = Kf + ((size_t)(b * SEQ) + kc + key) * DMODEL + h * HD;
      copy16_g2l(kr + d8, sK + key * 64 + d8);
      const _Float16* vr = Vf + ((size_t)(b * SEQ) + kc + key) * DMODEL + h * HD;
      v8h vv = *(const v8h*)(vr + d8);
#pragma unroll
      for (int j = 0; j < 8; ++j) sVt[(d8 + j) * 32 + key] = vv[j];
    }
    wait_async_lds();
    __syncthreads();

    // logits = Q @ K^T over 32 keys: two 16-key column tiles, 2 d-steps each
    v8f lg[2] = {};
#pragma unroll
    for (int s = 0; s < 2; ++s) {
#pragma unroll
      for (int nt = 0; nt < 2; ++nt) {
        v16h bk = load_fragB(sK + (size_t)(nt * 16) * 64 + s * 32, 64);
        lg[nt] = __builtin_amdgcn_wmma_f32_16x16x32_f16(
            false, aq[s], false, bk, (short)0, lg[nt], false, false);
      }
    }
    // scale + key mask (per column -> lane-local, mask from LDS)
#pragma unroll
    for (int nt = 0; nt < 2; ++nt) {
      bool ok = sMask[kc + nt * 16 + cn] != 0;
#pragma unroll
      for (int r = 0; r < 8; ++r)
        lg[nt][r] = ok ? lg[nt][r] * 0.125f : -1.0e30f;
    }

    // online softmax: row stats live across 16 lanes of each half-wave
    float cmax[8];
#pragma unroll
    for (int r = 0; r < 8; ++r) cmax[r] = fmaxf(lg[0][r], lg[1][r]);
#pragma unroll
    for (int msk = 8; msk >= 1; msk >>= 1)
#pragma unroll
      for (int r = 0; r < 8; ++r)
        cmax[r] = fmaxf(cmax[r], __shfl_xor(cmax[r], msk));

    float scale[8], csum[8];
    v8f p0, p1;
#pragma unroll
    for (int r = 0; r < 8; ++r) {
      float mnew = fmaxf(mrun[r], cmax[r]);
      scale[r] = __expf(mrun[r] - mnew);
      mrun[r]  = mnew;
      p0[r] = __expf(lg[0][r] - mnew);
      p1[r] = __expf(lg[1][r] - mnew);
      csum[r] = p0[r] + p1[r];
    }
#pragma unroll
    for (int msk = 8; msk >= 1; msk >>= 1)
#pragma unroll
      for (int r = 0; r < 8; ++r) csum[r] += __shfl_xor(csum[r], msk);
#pragma unroll
    for (int r = 0; r < 8; ++r) lrun[r] = lrun[r] * scale[r] + csum[r];
#pragma unroll
    for (int j = 0; j < 4; ++j)
#pragma unroll
      for (int r = 0; r < 8; ++r) acc[j][r] *= scale[r];

    // stage P (16q x 32keys) in this wave's private LDS region, reload as A frag
    _Float16* sPw = sP + wave * (16 * 32);
#pragma unroll
    for (int nt = 0; nt < 2; ++nt)
#pragma unroll
      for (int r = 0; r < 8; ++r)
        sPw[(r + 8 * hi) * 32 + nt * 16 + cn] =
            (_Float16)(nt ? p1[r] : p0[r]);
    asm volatile("s_wait_dscnt 0" ::: "memory");  // CDNA5 split DS counter

    v16h pf = load_fragA(sPw, 32);
#pragma unroll
    for (int j = 0; j < 4; ++j) {
      v16h bv = load_fragB(sVt + (size_t)(j * 16) * 32, 32);
      acc[j] = __builtin_amdgcn_wmma_f32_16x16x32_f16(
          false, pf, false, bv, (short)0, acc[j], false, false);
    }
  }

  // normalize and store attn output (f16, laid out [B*T, H*DV])
  float rl[8];
#pragma unroll
  for (int r = 0; r < 8; ++r) rl[r] = 1.0f / lrun[r];
#pragma unroll
  for (int j = 0; j < 4; ++j)
#pragma unroll
    for (int r = 0; r < 8; ++r) {
      size_t row = (size_t)(b * SEQ) + q0 + wave * 16 + r + 8 * hi;
      O[row * DMODEL + h * HD + j * 16 + cn] = (_Float16)(acc[j][r] * rl[r]);
    }
}

// ---------------------------------------------------------------------------
extern "C" void kernel_launch(void* const* d_in, const int* in_sizes, int n_in,
                              void* d_out, int out_size, void* d_ws, size_t ws_size,
                              hipStream_t stream) {
  (void)in_sizes; (void)n_in; (void)out_size; (void)ws_size;
  const float* x      = (const float*)d_in[0];
  const float* y      = (const float*)d_in[1];
  const unsigned char* mask = (const unsigned char*)d_in[2];
  const float* ln1_g  = (const float*)d_in[3];
  const float* ln1_b  = (const float*)d_in[4];
  const float* ln2_g  = (const float*)d_in[5];
  const float* ln2_b  = (const float*)d_in[6];
  const float* q_w = (const float*)d_in[7];   const float* q_b = (const float*)d_in[8];
  const float* k_w = (const float*)d_in[9];   const float* k_b = (const float*)d_in[10];
  const float* v_w = (const float*)d_in[11];  const float* v_b = (const float*)d_in[12];
  const float* o_w = (const float*)d_in[13];  const float* o_b = (const float*)d_in[14];
  const float* mln1_g = (const float*)d_in[15]; const float* mln1_b = (const float*)d_in[16];
  const float* l1_w = (const float*)d_in[17];   const float* l1_b = (const float*)d_in[18];
  const float* mln2_g = (const float*)d_in[19]; const float* mln2_b = (const float*)d_in[20];
  const float* l2_w = (const float*)d_in[21];   const float* l2_b = (const float*)d_in[22];

  char* ws = (char*)d_ws;
  size_t off = 0;
  auto carve = [&](size_t bytes) -> char* {
    char* p = ws + off;
    off += (bytes + 255) & ~(size_t)255;
    return p;
  };
  const size_t WB = (size_t)DMODEL * DMODEL * sizeof(_Float16);  // 2 MB
  const size_t AB = (size_t)NROWS * DMODEL * sizeof(_Float16);   // 8 MB
  const size_t FB = (size_t)NROWS * DMODEL * sizeof(float);      // 16 MB

  _Float16* qwf = (_Float16*)carve(WB);
  _Float16* kwf = (_Float16*)carve(WB);
  _Float16* vwf = (_Float16*)carve(WB);
  _Float16* owf = (_Float16*)carve(WB);
  _Float16* l1wf = (_Float16*)carve(WB);
  _Float16* l2wf = (_Float16*)carve(WB);
  _Float16* x1 = (_Float16*)carve(AB);
  _Float16* y1 = (_Float16*)carve(AB);
  _Float16* Qf = (_Float16*)carve(AB);
  _Float16* Kf = (_Float16*)carve(AB);
  _Float16* Vf = (_Float16*)carve(AB);
  float* xres = (float*)carve(FB);
  float* hmid = (float*)carve(FB);
  _Float16* attn = x1;  // x1 dead after Q projection
  _Float16* h1   = y1;  // y1 dead after K/V projections
  _Float16* h2   = Qf;  // Qf dead after attention

  dim3 blk(256);
  const int WN = DMODEL * DMODEL;
  f32_to_f16_kernel<<<dim3(512), blk, 0, stream>>>(q_w, qwf, WN);
  f32_to_f16_kernel<<<dim3(512), blk, 0, stream>>>(k_w, kwf, WN);
  f32_to_f16_kernel<<<dim3(512), blk, 0, stream>>>(v_w, vwf, WN);
  f32_to_f16_kernel<<<dim3(512), blk, 0, stream>>>(o_w, owf, WN);
  f32_to_f16_kernel<<<dim3(512), blk, 0, stream>>>(l1_w, l1wf, WN);
  f32_to_f16_kernel<<<dim3(512), blk, 0, stream>>>(l2_w, l2wf, WN);

  ln_relu_f16_kernel<<<dim3(NROWS), blk, 0, stream>>>(x, ln1_g, ln1_b, x1, DMODEL);
  ln_relu_f16_kernel<<<dim3(NROWS), blk, 0, stream>>>(y, ln2_g, ln2_b, y1, DMODEL);

  dim3 gg(NROWS / 128, DMODEL / 128);
  gemm_wmma_kernel<0><<<gg, blk, 0, stream>>>(x1, qwf, q_b, nullptr, nullptr, Qf,
                                              NROWS, DMODEL, DMODEL);
  gemm_wmma_kernel<0><<<gg, blk, 0, stream>>>(y1, kwf, k_b, nullptr, nullptr, Kf,
                                              NROWS, DMODEL, DMODEL);
  gemm_wmma_kernel<0><<<gg, blk, 0, stream>>>(y1, vwf, v_b, nullptr, nullptr, Vf,
                                              NROWS, DMODEL, DMODEL);

  attn_kernel<<<dim3(SEQ / 128, BATCH * NHEAD), blk, 0, stream>>>(Qf, Kf, Vf, mask, attn);

  gemm_wmma_kernel<1><<<gg, blk, 0, stream>>>(attn, owf, o_b, x, xres, nullptr,
                                              NROWS, DMODEL, DMODEL);

  ln_relu_f16_kernel<<<dim3(NROWS), blk, 0, stream>>>(xres, mln1_g, mln1_b, h1, DMODEL);
  gemm_wmma_kernel<2><<<gg, blk, 0, stream>>>(h1, l1wf, l1_b, nullptr, hmid, nullptr,
                                              NROWS, DMODEL, DMODEL);
  ln_relu_f16_kernel<<<dim3(NROWS), blk, 0, stream>>>(hmid, mln2_g, mln2_b, h2, DMODEL);
  gemm_wmma_kernel<2><<<gg, blk, 0, stream>>>(h2, l2wf, l2_b, nullptr, (float*)d_out, nullptr,
                                              NROWS, DMODEL, DMODEL);
}
